// Message_53798760349729
// MI455X (gfx1250) — matrix operand order repaired and checked by reading, not compiled, not run on hardware
//
#include <hip/hip_runtime.h>

// ---------------------------------------------------------------------------
// PaiNN message block, MI455X (gfx1250, wave32, WMMA).
//  h    = silu(s_j @ W1^T + b1)                (E,128)   WMMA f16->f32
//  phi  = h @ W2^T + b2                        (E,384)   WMMA f16->f32
//  rbf  = sin(n*pi/5*|r|)/|r|  (K padded 20->32)
//  rout = rbf @ Wr^T + br                      (E,384)   WMMA f16->f32
//  conv = phi * fc * rout ; split into (vj-gate, rj-gate, ds)
//  segment-sum via global_atomic_add_f32 (outputs live in L2: 41MB << 192MB)
// Roofline: ~59 GFLOP f16-WMMA vs ~870 MB @ 23.3 TB/s (~37us floor) -> balanced.
// GEMM2 is ordered kk-outer/j-inner: one A fragment feeds 3 WMMAs on 3
// independent accumulators (1 dscnt wait per 3 WMMAs, pipelined matrix unit).
// ---------------------------------------------------------------------------

#define FDIM 128
#define NRBF 20
#define ME   16     // edges per tile (one WMMA M)
#define PI_F 3.14159265358979323846f

typedef __attribute__((ext_vector_type(16))) _Float16 v16h;
typedef __attribute__((ext_vector_type(8)))  _Float16 v8h;
typedef __attribute__((ext_vector_type(8)))  float    v8f;

__device__ __forceinline__ v8f wmma16(v16h a, v16h b, v8f c) {
  // v_wmma_f32_16x16x32_f16
  return __builtin_amdgcn_wmma_f32_16x16x32_f16(
      /*neg_a=*/false, a, /*neg_b=*/false, b,
      /*c_mod=*/(short)0, c, /*reuse_a=*/false, /*reuse_b=*/false);
}

// B-fragment (32x16, f16) for D = A x B with B[k][n] = W[n][k], W f32 row-major.
// ISA B layout: lane half h -> N=lane%16, K = k0+{h*8+0..7} U {k0+16+h*8+0..7} (matches A).
__device__ __forceinline__ v16h bfrag_f32(const float* __restrict__ W, int ld,
                                          int n0, int k0, int lane) {
  const int n  = n0 + (lane & 15);
  const int kb = k0 + ((lane >> 4) << 3);
  const float4* p = (const float4*)(W + (size_t)n * ld + kb);
  float4 a0 = p[0], a1 = p[1], a2 = p[4], a3 = p[5];
  v16h b;
  b[0]  = (_Float16)a0.x; b[1]  = (_Float16)a0.y; b[2]  = (_Float16)a0.z; b[3]  = (_Float16)a0.w;
  b[4]  = (_Float16)a1.x; b[5]  = (_Float16)a1.y; b[6]  = (_Float16)a1.z; b[7]  = (_Float16)a1.w;
  b[8]  = (_Float16)a2.x; b[9]  = (_Float16)a2.y; b[10] = (_Float16)a2.z; b[11] = (_Float16)a2.w;
  b[12] = (_Float16)a3.x; b[13] = (_Float16)a3.y; b[14] = (_Float16)a3.z; b[15] = (_Float16)a3.w;
  return b;
}

// Same but K dimension clamped to NRBF (Wr is (384,20)); pad with zeros.
__device__ __forceinline__ v16h bfrag_wr(const float* __restrict__ Wr, int n0, int lane) {
  const int n  = n0 + (lane & 15);
  const int kb = (lane >> 4) << 3;
  v16h b;
#pragma unroll
  for (int i = 0; i < 16; ++i) {
    const int k = kb + (i < 8 ? i : i + 8);
    b[i] = (k < NRBF) ? (_Float16)Wr[(size_t)n * NRBF + k] : (_Float16)0.0f;
  }
  return b;
}

// A-fragment (16x32, f16) from LDS row-major [16][ldk] at K offset k0.
// ISA A layout: lane half selects K = {h*8+0..7} U {16+h*8+0..7}  -> two b128 DS reads.
__device__ __forceinline__ v16h afrag_lds(const _Float16* s, int ldk, int k0, int lane) {
  const int m  = lane & 15;
  const int h8 = (lane >> 4) << 3;   // 0 or 8
  const _Float16* p = s + m * ldk + k0 + h8;
  v8h lo = *(const v8h*)p;
  v8h hi = *(const v8h*)(p + 16);
  return __builtin_shufflevector(lo, hi, 0,1,2,3,4,5,6,7,8,9,10,11,12,13,14,15);
}

__global__ void zero_kernel(float* __restrict__ p, long long n) {
  long long i = (long long)blockIdx.x * blockDim.x + threadIdx.x;
  const long long stride = (long long)gridDim.x * blockDim.x;
  const long long n4 = n >> 2;
  float4* p4 = (float4*)p;
  for (long long j = i; j < n4; j += stride) p4[j] = make_float4(0.f, 0.f, 0.f, 0.f);
  for (long long j = (n4 << 2) + i; j < n; j += stride) p[j] = 0.f;
}

__global__ __launch_bounds__(256, 1)
void painn_msg_kernel(const float* __restrict__ v_j, const float* __restrict__ s_j,
                      const float* __restrict__ r_ij, const int* __restrict__ A_row,
                      const float* __restrict__ W1, const float* __restrict__ b1,
                      const float* __restrict__ W2, const float* __restrict__ b2,
                      const float* __restrict__ Wr, const float* __restrict__ br,
                      float* __restrict__ out_v,   // (N,F,3)
                      float* __restrict__ out_s,   // (N,F)
                      int E, int numTiles) {
  __shared__ __align__(16) _Float16 sS[ME * FDIM];   // s_j tile (f16)
  __shared__ __align__(16) _Float16 sH[ME * FDIM];   // h tile (f16)
  __shared__ __align__(16) _Float16 sRBF[ME * 32];   // rbf tile, K padded to 32
  __shared__ float sRU[ME][3];
  __shared__ float sFC[ME];
  __shared__ float sRN[ME];
  __shared__ float sINV[ME];
  __shared__ int   sRow[ME];

  const int tid  = threadIdx.x;
  const int lane = tid & 31;
  const int wave = tid >> 5;          // 0..7: this wave's N-tile group

  // ------- hoisted loop-invariant weight fragments (live in VGPRs) -------
  v16h w1f[4];
#pragma unroll
  for (int kk = 0; kk < 4; ++kk)
    w1f[kk] = bfrag_f32(W1, FDIM, wave * 16, kk * 32, lane);

  v16h w2f[4][3];   // [kk][j]
#pragma unroll
  for (int kk = 0; kk < 4; ++kk)
#pragma unroll
    for (int j = 0; j < 3; ++j)
      w2f[kk][j] = bfrag_f32(W2, FDIM, (wave + 8 * j) * 16, kk * 32, lane);

  v16h wrf[3];
#pragma unroll
  for (int j = 0; j < 3; ++j)
    wrf[j] = bfrag_wr(Wr, (wave + 8 * j) * 16, lane);

  const int   ncol = wave * 16 + (lane & 15);   // feature column in [0,128)
  const float b1v  = b1[ncol];
  float b2v[3], brv[3];
#pragma unroll
  for (int j = 0; j < 3; ++j) {
    b2v[j] = b2[(wave + 8 * j) * 16 + (lane & 15)];
    brv[j] = br[(wave + 8 * j) * 16 + (lane & 15)];
  }

  for (int t = blockIdx.x; t < numTiles; t += gridDim.x) {
    const int e0 = t * ME;

    // ---- stage s_j tile: 256 threads x 8 f32 -> f16 LDS ----
    {
      const int r = tid >> 4;
      const int c = (tid & 15) << 3;
      int e = e0 + r; if (e >= E) e = E - 1;
      const float4* src = (const float4*)(s_j + (size_t)e * FDIM + c);
      float4 x0 = src[0], x1 = src[1];
      _Float16* dst = sS + r * FDIM + c;
      dst[0] = (_Float16)x0.x; dst[1] = (_Float16)x0.y;
      dst[2] = (_Float16)x0.z; dst[3] = (_Float16)x0.w;
      dst[4] = (_Float16)x1.x; dst[5] = (_Float16)x1.y;
      dst[6] = (_Float16)x1.z; dst[7] = (_Float16)x1.w;
    }
    // ---- per-edge geometry (exact 1/rn once; reused by 320 rbf slots) ----
    if (tid < ME) {
      int e = e0 + tid; if (e >= E) e = E - 1;
      const float rx = r_ij[(size_t)e * 3 + 0];
      const float ry = r_ij[(size_t)e * 3 + 1];
      const float rz = r_ij[(size_t)e * 3 + 2];
      const float rn  = sqrtf(rx * rx + ry * ry + rz * rz);
      const float inv = 1.0f / rn;
      sRU[tid][0] = rx * inv; sRU[tid][1] = ry * inv; sRU[tid][2] = rz * inv;
      sRN[tid]  = rn;
      sINV[tid] = inv;
      sFC[tid]  = 0.5f * (__cosf(PI_F * rn * 0.2f) + 1.0f);   // v_cos_f32
      sRow[tid] = A_row[e];
    }
    __syncthreads();

    // ---- rbf tile: exactly 2 slots per thread (512 = 2*256), no loop ----
    {
      const int m0 = tid >> 5,        k0 = tid & 31;          // slot tid
      const int m1 = (tid + 256) >> 5, k1 = tid & 31;         // slot tid+256
      const float v0 = (k0 < NRBF)
          ? __sinf((float)(k0 + 1) * (PI_F * 0.2f) * sRN[m0]) * sINV[m0] : 0.0f;
      const float v1 = (k1 < NRBF)
          ? __sinf((float)(k1 + 1) * (PI_F * 0.2f) * sRN[m1]) * sINV[m1] : 0.0f;
      sRBF[tid]       = (_Float16)v0;
      sRBF[tid + 256] = (_Float16)v1;
    }

    // ---- GEMM1: h = silu(s W1^T + b1); two independent WMMA chains ----
    v16h a0 = afrag_lds(sS, FDIM, 0,  lane);
    v16h a1 = afrag_lds(sS, FDIM, 32, lane);
    v16h a2 = afrag_lds(sS, FDIM, 64, lane);
    v16h a3 = afrag_lds(sS, FDIM, 96, lane);
    v8f accA = {}, accB = {};
    accA = wmma16(a0, w1f[0], accA);
    accB = wmma16(a1, w1f[1], accB);
    accA = wmma16(a2, w1f[2], accA);
    accB = wmma16(a3, w1f[3], accB);
    {
      const int mb = (lane >> 4) << 3;
#pragma unroll
      for (int r = 0; r < 8; ++r) {
        const float x = (accA[r] + accB[r]) + b1v;
        // silu via native v_exp_f32 + v_rcp_f32 (no IEEE divide chain)
        const float h = x * __builtin_amdgcn_rcpf(1.0f + __expf(-x));
        sH[(mb + r) * FDIM + ncol] = (_Float16)h;
      }
    }
    __syncthreads();

    // ---- GEMM3 (rbf_out) first: its 3 WMMAs cover the first sH load ----
    v8f accp[3] = {}, accr[3] = {};
    {
      v16h ar = afrag_lds(sRBF, 32, 0, lane);
      v16h a  = afrag_lds(sH, FDIM, 0, lane);      // issue early
      accr[0] = wmma16(ar, wrf[0], accr[0]);
      accr[1] = wmma16(ar, wrf[1], accr[1]);
      accr[2] = wmma16(ar, wrf[2], accr[2]);
      // ---- GEMM2: kk-outer / j-inner, software-pipelined A fragments ----
#pragma unroll
      for (int kk = 0; kk < 4; ++kk) {
        v16h anext = (kk < 3) ? afrag_lds(sH, FDIM, (kk + 1) * 32, lane) : a;
        accp[0] = wmma16(a, w2f[kk][0], accp[0]);
        accp[1] = wmma16(a, w2f[kk][1], accp[1]);
        accp[2] = wmma16(a, w2f[kk][2], accp[2]);
        a = anext;
      }
    }

    // ---- epilogue: conv = phi * fc * rout ; scatter-add ----
    {
      const int  mb   = (lane >> 4) << 3;
      const bool full = (e0 + ME) <= E;
#pragma unroll
      for (int r = 0; r < 8; ++r) {
        const int m = mb + r;
        const int e = e0 + m;
        if (full || e < E) {
          const float fc  = sFC[m];
          const int   row = sRow[m];
          const float cv = (accp[0][r] + b2v[0]) * fc * (accr[0][r] + brv[0]);
          const float cr = (accp[1][r] + b2v[1]) * fc * (accr[1][r] + brv[1]);
          const float cs = (accp[2][r] + b2v[2]) * fc * (accr[2][r] + brv[2]);
          const float* vj = v_j + ((size_t)e * FDIM + ncol) * 3;
          float* ov = out_v + ((size_t)row * FDIM + ncol) * 3;
          atomicAdd(ov + 0, vj[0] * cv + cr * sRU[m][0]);
          atomicAdd(ov + 1, vj[1] * cv + cr * sRU[m][1]);
          atomicAdd(ov + 2, vj[2] * cv + cr * sRU[m][2]);
          atomicAdd(out_s + (size_t)row * FDIM + ncol, cs);
        }
      }
    }
    __syncthreads();   // protect LDS tiles before next iteration
  }
}

extern "C" void kernel_launch(void* const* d_in, const int* in_sizes, int n_in,
                              void* d_out, int out_size, void* d_ws, size_t ws_size,
                              hipStream_t stream) {
  const float* v_j   = (const float*)d_in[0];
  const float* s_j   = (const float*)d_in[1];
  const float* r_ij  = (const float*)d_in[2];
  const int*   A_row = (const int*)d_in[4];
  const float* W1    = (const float*)d_in[5];
  const float* b1    = (const float*)d_in[6];
  const float* W2    = (const float*)d_in[7];
  const float* b2    = (const float*)d_in[8];
  const float* Wr    = (const float*)d_in[9];
  const float* br    = (const float*)d_in[10];

  const int E = in_sizes[1] / FDIM;           // 400000
  const int N = out_size / (4 * FDIM);        // 20000 (out = N*F*3 + N*F)
  float* out_v = (float*)d_out;
  float* out_s = out_v + (size_t)N * FDIM * 3;

  // outputs accumulate via atomics -> zero every launch
  zero_kernel<<<2048, 256, 0, stream>>>((float*)d_out, (long long)out_size);

  const int numTiles = (E + ME - 1) / ME;
  const int grid = numTiles < 2048 ? numTiles : 2048;
  painn_msg_kernel<<<grid, 256, 0, stream>>>(v_j, s_j, r_ij, A_row,
                                             W1, b1, W2, b2, Wr, br,
                                             out_v, out_s, E, numTiles);
}
